// Attention_3470333575830
// MI455X (gfx1250) — matrix operand order repaired and checked by reading, not compiled, not run on hardware
//
#include <hip/hip_runtime.h>

typedef __bf16 bf16;
typedef __attribute__((ext_vector_type(16))) __bf16 v16bf;
typedef __attribute__((ext_vector_type(8)))  float  v8f;

// Address-space(3) typed LDS pointers: guarantees ds_load/ds_store codegen
// (generic pointers were lowering to flat_load in loop bodies).
typedef __attribute__((address_space(3))) bf16 lds_bf16;

constexpr int kB = 4, kT = 2048, kC = 1024, kH = 16, kD = 64;
constexpr int kM = kB * kT;          // 8192 rows
constexpr int kNqkv = 3 * kC;        // 3072

// ---------------------------------------------------------------------------
// CDNA5 async global->LDS path (guarded; signature probed in round 2/3:
// (v4i addrspace(1)* src, v4i addrspace(3)* dst, imm offset, imm cpol)).
// ---------------------------------------------------------------------------
#if defined(__AMDGCN__) &&                                                     \
    __has_builtin(__builtin_amdgcn_global_load_async_to_lds_b128) &&           \
    __has_builtin(__builtin_amdgcn_s_wait_asynccnt)
#define HAVE_ASYNC 1
typedef int v4i __attribute__((ext_vector_type(4)));
typedef __attribute__((address_space(1))) v4i gas_v4i;
typedef __attribute__((address_space(3))) v4i las_v4i;
#else
#define HAVE_ASYNC 0
#endif

// Each stage_begin/stage_end pair moves 32 bytes (16 bf16) per thread.
struct Stage { uint4 a, b; };

__device__ __forceinline__ Stage stage_begin(bf16* l, const bf16* g) {
  Stage s = {};
#if HAVE_ASYNC
  __builtin_amdgcn_global_load_async_to_lds_b128((gas_v4i*)g, (las_v4i*)l,
                                                 0, 0);
  __builtin_amdgcn_global_load_async_to_lds_b128((gas_v4i*)g, (las_v4i*)l,
                                                 16, 0);
#else
  s.a = ((const uint4*)g)[0];
  s.b = ((const uint4*)g)[1];
  (void)l;
#endif
  return s;
}

__device__ __forceinline__ void stage_end(bf16* l, const Stage& s) {
#if !HAVE_ASYNC
  ((uint4*)l)[0] = s.a;
  ((uint4*)l)[1] = s.b;
#else
  (void)l; (void)s;
#endif
}

__device__ __forceinline__ void stage_wait_and_barrier() {
#if HAVE_ASYNC
  __builtin_amdgcn_s_wait_asynccnt(0);
#endif
  __syncthreads();
}

// ---------------------------------------------------------------------------
// WMMA helpers (CDNA5 wave32, 16x16x32 bf16 -> f32)
// ---------------------------------------------------------------------------
__device__ __forceinline__ v8f wmma_bf16(v16bf a, v16bf b, v8f c) {
  return __builtin_amdgcn_wmma_f32_16x16x32_bf16(
      false, a, false, b, (short)0, c, false, false);
}

// A-matrix 16x32 bf16 (ISA 7.12.2): lane holds row m=lane%16; lane-half picks
// k-subranges (v<4 -> half*8+2v, v>=4 -> 16+half*8+2(v-4)).
__device__ __forceinline__ v16bf load_a16x32(const bf16* p, int ld) {
  const int lane = threadIdx.x & 31;
  const int m = lane & 15, lh = lane >> 4;
  const bf16* row = p + m * ld + lh * 8;
  v16bf a;
#pragma unroll
  for (int v = 0; v < 8; ++v) {
    const int k = (v < 4) ? (2 * v) : (16 + 2 * (v - 4));
    a[2 * v]     = row[k];
    a[2 * v + 1] = row[k + 1];
  }
  return a;
}

__device__ __forceinline__ v16bf load_a16x32_lds(const lds_bf16* p, int ld) {
  const int lane = threadIdx.x & 31;
  const int m = lane & 15, lh = lane >> 4;
  const lds_bf16* row = p + m * ld + lh * 8;
  v16bf a;
#pragma unroll
  for (int v = 0; v < 8; ++v) {
    const int k = (v < 4) ? (2 * v) : (16 + 2 * (v - 4));
    a[2 * v]     = row[k];
    a[2 * v + 1] = row[k + 1];
  }
  return a;
}

// B-matrix 32x16 bf16 from a transposed operand pT[n][k] (ld): lane holds
// column n=lane%16, lane-half picks k half*8+0..7 and 16+half*8+0..7.
__device__ __forceinline__ v16bf load_b32x16_lds(const lds_bf16* pT, int ld) {
  const int lane = threadIdx.x & 31;
  const int n = lane & 15, lh = lane >> 4;
  const lds_bf16* row = pT + n * ld + lh * 8;
  v16bf b;
#pragma unroll
  for (int v = 0; v < 16; ++v) b[v] = row[v < 8 ? v : v + 8];
  return b;
}

// ---------------------------------------------------------------------------
// Stage 0: f32 -> bf16 conversions
// ---------------------------------------------------------------------------
__global__ void cvt_bf16_kernel(bf16* __restrict__ dst,
                                const float* __restrict__ src, int n) {
  int i = blockIdx.x * blockDim.x + threadIdx.x;
  if (i < n) dst[i] = (bf16)src[i];
}

__global__ void transpose_cvt_kernel(bf16* __restrict__ dst,
                                     const float* __restrict__ src,
                                     int K, int N) {
  int i = blockIdx.x * blockDim.x + threadIdx.x;
  if (i < K * N) {
    int k = i / N, n = i % N;
    dst[(size_t)n * K + k] = (bf16)src[i];
  }
}

// ---------------------------------------------------------------------------
// Stages 1 & 3 core: C[128x128 block] = A[Mxk] @ B^T[Nxk], bf16 WMMA, f32 acc.
// 8 waves; wave tile 32(M) x 64(N); LDS double-buffered 128x32 A / B tiles.
// ---------------------------------------------------------------------------
struct Acc8 { v8f a[2][4]; };

constexpr int kTile = 128 * 32;      // one LDS tile (elements)

__device__ __forceinline__ Acc8 gemm_block_128x128(
    const bf16* __restrict__ Aglob,   // [*, kC] row-major, row m0blk
    const bf16* __restrict__ BTglob,  // [*, kC] row-major, row n0blk
    bf16* AsBase, bf16* BsBase) {     // LDS: 2 buffers of 128*32 each
  const int tid = threadIdx.x;
  const int wid = tid >> 5;
  const int wm = wid >> 1;            // 0..3  -> M offset wm*32
  const int wn = wid & 1;             // 0..1  -> N offset wn*64

  const lds_bf16* Asl = (const lds_bf16*)AsBase;
  const lds_bf16* Bsl = (const lds_bf16*)BsBase;

  const int base = tid * 16;          // 32B slice per thread per tile
  const int row = base >> 5;          // 0..127
  const int col = base & 31;          // 0 or 16

  Acc8 acc;
#pragma unroll
  for (int s = 0; s < 2; ++s)
#pragma unroll
    for (int j = 0; j < 4; ++j) acc.a[s][j] = v8f{};

  constexpr int nk = kC / 32;
  // Prologue: stage tile 0 into buffer 0.
  {
    Stage sa = stage_begin(AsBase + base, Aglob + (size_t)row * kC + col);
    Stage sb = stage_begin(BsBase + base, BTglob + (size_t)row * kC + col);
    stage_end(AsBase + base, sa);
    stage_end(BsBase + base, sb);
  }
  for (int i = 0; i < nk; ++i) {
    stage_wait_and_barrier();         // buf[cur] ready & visible
    const int cur = i & 1, nxt = cur ^ 1;
    Stage sa = {}, sb = {};
    const bool hasNext = (i + 1 < nk);
    if (hasNext) {
      const int kb = (i + 1) * 32;
      sa = stage_begin(AsBase + nxt * kTile + base,
                       Aglob + (size_t)row * kC + kb + col);
      sb = stage_begin(BsBase + nxt * kTile + base,
                       BTglob + (size_t)row * kC + kb + col);
    }
    // Compute from buf[cur] (integer offset keeps the LDS base constant so
    // the backend emits ds_load_b128, not flat_load).
    const int co = cur * kTile;
    v16bf a0 = load_a16x32_lds(Asl + co + (wm * 32) * 32, 32);
    v16bf a1 = load_a16x32_lds(Asl + co + (wm * 32 + 16) * 32, 32);
#pragma unroll
    for (int j = 0; j < 4; ++j) {
      v16bf b = load_b32x16_lds(Bsl + co + (wn * 64 + j * 16) * 32, 32);
      acc.a[0][j] = wmma_bf16(a0, b, acc.a[0][j]);
      acc.a[1][j] = wmma_bf16(a1, b, acc.a[1][j]);
    }
    __syncthreads();                  // everyone done reading buf[nxt]'s old data
    if (hasNext) {
      stage_end(AsBase + nxt * kTile + base, sa);
      stage_end(BsBase + nxt * kTile + base, sb);
    }
  }
  return acc;
}

// ---------------------------------------------------------------------------
// Stage 1: qkv GEMM scattering into head-major Q/K and transposed V.
// ---------------------------------------------------------------------------
__global__ __launch_bounds__(256)
void qkv_gemm_kernel(const bf16* __restrict__ xb, const bf16* __restrict__ wT,
                     bf16* __restrict__ Qh, bf16* __restrict__ Kh,
                     bf16* __restrict__ Vt) {
  __shared__ __align__(16) bf16 As[2 * kTile];
  __shared__ __align__(16) bf16 Bs[2 * kTile];
  const int m0 = blockIdx.x * 128;
  const int n0 = blockIdx.y * 128;
  Acc8 acc = gemm_block_128x128(xb + (size_t)m0 * kC, wT + (size_t)n0 * kC,
                                As, Bs);

  const int lane = threadIdx.x & 31;
  const int wid = threadIdx.x >> 5;
  const int wm = wid >> 1, wn = wid & 1;
  const int lh = lane >> 4, ncol = lane & 15;
  const int nwave = n0 + wn * 64;     // 64-aligned -> one of Q/K/V, one head
  const int which = nwave / kC;
  const int nin = nwave % kC;
  const int h = nin / kD;
  const int b_ = m0 / kT;
  const int t0 = m0 % kT + wm * 32;
  const size_t bh = (size_t)b_ * kH + h;
#pragma unroll
  for (int s = 0; s < 2; ++s)
#pragma unroll
    for (int j = 0; j < 4; ++j)
#pragma unroll
      for (int v = 0; v < 8; ++v) {
        const int tt = t0 + s * 16 + v + 8 * lh;
        const int d = (nin % kD) + j * 16 + ncol;
        const float val = acc.a[s][j][v];
        if (which == 0)
          Qh[(bh * kT + tt) * kD + d] = (bf16)(val * 0.125f); // 1/sqrt(64)
        else if (which == 1)
          Kh[(bh * kT + tt) * kD + d] = (bf16)val;
        else
          Vt[(bh * kD + d) * kT + tt] = (bf16)val;            // transposed
      }
}

// ---------------------------------------------------------------------------
// Stage 2: causal flash attention. 4 waves = 64 q rows of one (b,h);
// key loop step 32, K/V tiles double-buffered in LDS.
// ---------------------------------------------------------------------------
__global__ __launch_bounds__(128)
void flash_attn_kernel(const bf16* __restrict__ Qh, const bf16* __restrict__ Kh,
                       const bf16* __restrict__ Vt, bf16* __restrict__ att) {
  __shared__ __align__(16) bf16 Ktile[2][32 * 64];  // [key][d]
  __shared__ __align__(16) bf16 Vtile[2][64 * 32];  // [d][key]
  __shared__ __align__(16) bf16 Ptile[4][16 * 32];  // per-wave P transpose

  const int tid = threadIdx.x;
  const int wid = tid >> 5, lane = tid & 31;
  const int lh = lane >> 4, ncol = lane & 15;
  const int bh = blockIdx.y;
  const int qb0 = blockIdx.x * 64;
  const int q0 = qb0 + wid * 16;

  const bf16* Qbase = Qh + (size_t)bh * kT * kD;
  const bf16* Kbase = Kh + (size_t)bh * kT * kD;
  const bf16* Vbase = Vt + (size_t)bh * kD * kT;

  const lds_bf16* Kl = (const lds_bf16*)&Ktile[0][0];
  const lds_bf16* Vl = (const lds_bf16*)&Vtile[0][0];
  lds_bf16* Pl = (lds_bf16*)&Ptile[0][0];

  v16bf aq0 = load_a16x32(Qbase + (size_t)q0 * kD + 0, kD);
  v16bf aq1 = load_a16x32(Qbase + (size_t)q0 * kD + 32, kD);

  float rm[8], rl[8];
  v8f o[4] = {};
#pragma unroll
  for (int v = 0; v < 8; ++v) { rm[v] = -1e30f; rl[v] = 0.0f; }

  // Per-thread staging slices.
  const int base = tid * 16;
  const int krow = base >> 6, kcol = base & 63;   // K tile: 32x64
  const int vrow = base >> 5, vcol = base & 31;   // V tile: 64x32

  const int kend = qb0 + 64;          // causal bound, uniform per block
  const int nk = kend / 32;
  {
    Stage sk = stage_begin(&Ktile[0][base],
                           Kbase + (size_t)krow * kD + kcol);
    Stage sv = stage_begin(&Vtile[0][base],
                           Vbase + (size_t)vrow * kT + vcol);
    stage_end(&Ktile[0][base], sk);
    stage_end(&Vtile[0][base], sv);
  }
  for (int i = 0; i < nk; ++i) {
    stage_wait_and_barrier();
    const int cur = i & 1, nxt = cur ^ 1;
    const int kt = i * 32;
    Stage sk = {}, sv = {};
    const bool hasNext = (i + 1 < nk);
    if (hasNext) {
      const int ktn = kt + 32;
      sk = stage_begin(&Ktile[nxt][base],
                       Kbase + (size_t)(ktn + krow) * kD + kcol);
      sv = stage_begin(&Vtile[nxt][base],
                       Vbase + (size_t)vrow * kT + ktn + vcol);
    }

    // S = Q @ K^T (two 16-key groups, d split into 2 WMMAs each)
    const int kco = cur * (32 * 64);
    v8f s[2] = {};
#pragma unroll
    for (int g = 0; g < 2; ++g) {
      s[g] = wmma_bf16(aq0, load_b32x16_lds(Kl + kco + g * 16 * 64 + 0, 64), s[g]);
      s[g] = wmma_bf16(aq1, load_b32x16_lds(Kl + kco + g * 16 * 64 + 32, 64), s[g]);
    }

    // Causal mask + online softmax (row lives across 16 lanes of this half).
    float pr[2][8];
#pragma unroll
    for (int g = 0; g < 2; ++g)
#pragma unroll
      for (int v = 0; v < 8; ++v) {
        const int m = v + 8 * lh;
        const int key = kt + g * 16 + ncol;
        pr[g][v] = (key <= q0 + m) ? s[g][v] : -1e30f;
      }
#pragma unroll
    for (int v = 0; v < 8; ++v) {
      float mx = fmaxf(pr[0][v], pr[1][v]);
#pragma unroll
      for (int off = 1; off < 16; off <<= 1)
        mx = fmaxf(mx, __shfl_xor(mx, off, 32));
      const float mnew = fmaxf(rm[v], mx);
      const float alpha = __expf(rm[v] - mnew);
      rm[v] = mnew;
      float psum = 0.0f;
#pragma unroll
      for (int g = 0; g < 2; ++g) {
        const float p = __expf(pr[g][v] - mnew);
        pr[g][v] = p;
        psum += p;
      }
#pragma unroll
      for (int off = 1; off < 16; off <<= 1)
        psum += __shfl_xor(psum, off, 32);
      rl[v] = rl[v] * alpha + psum;
#pragma unroll
      for (int g = 0; g < 4; ++g) o[g][v] *= alpha;
    }

    // P: C-layout -> A-layout via per-wave LDS (per-wave DS ops are in-order).
    lds_bf16* pt = Pl + wid * (16 * 32);
#pragma unroll
    for (int g = 0; g < 2; ++g)
#pragma unroll
      for (int v = 0; v < 8; ++v)
        pt[(v + 8 * lh) * 32 + g * 16 + ncol] = (bf16)pr[g][v];
    v16bf ap = load_a16x32_lds(pt, 32);

    // O += P @ V
    const int vco = cur * (64 * 32);
#pragma unroll
    for (int g = 0; g < 4; ++g)
      o[g] = wmma_bf16(ap, load_b32x16_lds(Vl + vco + g * 16 * 32, 32), o[g]);

    __syncthreads();
    if (hasNext) {
      stage_end(&Ktile[nxt][base], sk);
      stage_end(&Vtile[nxt][base], sv);
    }
  }

  const int b_ = bh / kH, h = bh % kH;
#pragma unroll
  for (int g = 0; g < 4; ++g)
#pragma unroll
    for (int v = 0; v < 8; ++v) {
      const int tt = q0 + v + 8 * lh;
      const int d = g * 16 + ncol;
      att[((size_t)b_ * kT + tt) * kC + h * kD + d] = (bf16)(o[g][v] / rl[v]);
    }
}

// ---------------------------------------------------------------------------
// Stage 3: out = att @ w_proj (f32 straight to d_out)
// ---------------------------------------------------------------------------
__global__ __launch_bounds__(256)
void proj_gemm_kernel(const bf16* __restrict__ att, const bf16* __restrict__ wT,
                      float* __restrict__ out) {
  __shared__ __align__(16) bf16 As[2 * kTile];
  __shared__ __align__(16) bf16 Bs[2 * kTile];
  const int m0 = blockIdx.x * 128;
  const int n0 = blockIdx.y * 128;
  Acc8 acc = gemm_block_128x128(att + (size_t)m0 * kC, wT + (size_t)n0 * kC,
                                As, Bs);
  const int lane = threadIdx.x & 31;
  const int wid = threadIdx.x >> 5;
  const int wm = wid >> 1, wn = wid & 1;
  const int lh = lane >> 4, ncol = lane & 15;
#pragma unroll
  for (int s = 0; s < 2; ++s)
#pragma unroll
    for (int j = 0; j < 4; ++j)
#pragma unroll
      for (int v = 0; v < 8; ++v)
        out[(size_t)(m0 + wm * 32 + s * 16 + v + 8 * lh) * kC +
            n0 + wn * 64 + j * 16 + ncol] = acc.a[s][j][v];
}

// ---------------------------------------------------------------------------
extern "C" void kernel_launch(void* const* d_in, const int* in_sizes, int n_in,
                              void* d_out, int out_size, void* d_ws,
                              size_t ws_size, hipStream_t stream) {
  const float* x      = (const float*)d_in[0];   // [B,T,C]
  const float* w_qkv  = (const float*)d_in[1];   // [C,3C]
  const float* w_proj = (const float*)d_in[2];   // [C,C]
  float* out = (float*)d_out;                    // [B,T,C] f32

  bf16* xb     = (bf16*)d_ws;                       // 8192*1024
  bf16* wqkvT  = xb + (size_t)kM * kC;              // [n][k] 3072x1024
  bf16* wprojT = wqkvT + (size_t)kNqkv * kC;        // [n][k] 1024x1024
  bf16* Qh     = wprojT + (size_t)kC * kC;          // [b][h][t][d] (pre-scaled)
  bf16* Kh     = Qh + (size_t)kB * kH * kT * kD;    // [b][h][t][d]
  bf16* Vt     = Kh + (size_t)kB * kH * kT * kD;    // [b][h][d][t]
  bf16* attb   = Vt + (size_t)kB * kH * kT * kD;    // [b][t][h*d]

  {
    int n = kM * kC;
    cvt_bf16_kernel<<<(n + 255) / 256, 256, 0, stream>>>(xb, x, n);
    int nw = kC * kNqkv;
    transpose_cvt_kernel<<<(nw + 255) / 256, 256, 0, stream>>>(wqkvT, w_qkv,
                                                               kC, kNqkv);
    int np = kC * kC;
    transpose_cvt_kernel<<<(np + 255) / 256, 256, 0, stream>>>(wprojT, w_proj,
                                                               kC, kC);
  }
  qkv_gemm_kernel<<<dim3(kM / 128, kNqkv / 128), 256, 0, stream>>>(
      xb, wqkvT, Qh, Kh, Vt);
  flash_attn_kernel<<<dim3(kT / 64, kB * kH), 128, 0, stream>>>(Qh, Kh, Vt,
                                                                attb);
  proj_gemm_kernel<<<dim3(kM / 128, kC / 128), 256, 0, stream>>>(attb, wprojT,
                                                                 out);
}
